// MDCSACRF_83219286327627
// MI455X (gfx1250) — matrix operand order repaired and checked by reading, not compiled, not run on hardware
//
#include <hip/hip_runtime.h>
#include <hip/hip_bf16.h>

typedef __attribute__((ext_vector_type(16))) _Float16 v16h;
typedef __attribute__((ext_vector_type(8)))  _Float16 v8h;
typedef __attribute__((ext_vector_type(4)))  _Float16 v4h;
typedef __attribute__((ext_vector_type(2)))  _Float16 v2h;
typedef __attribute__((ext_vector_type(8)))  float    v8f;
typedef __attribute__((ext_vector_type(4)))  float    v4f;

constexpr int cB = 64, cT = 512, cH = 512, cHEADS = 4, cL = 16, cR = 256, cA = 9;
constexpr int cDH = 128, cF = 2048;
constexpr int cM = cB * cT;                 // 32768 rows
constexpr size_t cNE = (size_t)cM * cH;     // elements in one (B,T,H) activation

// ---------------------------------------------------------------------------
// WMMA helpers (CDNA5 gfx1250, wave32)
// ---------------------------------------------------------------------------
__device__ __forceinline__ v8f wmma_f16f32(v16h a, v16h b, v8f c) {
  return __builtin_amdgcn_wmma_f32_16x16x32_f16(false, a, false, b, (short)0, c,
                                                false, false);
}

// A fragment: 16x32 f16. Lane holds row M=lane&15.
__device__ __forceinline__ v16h load_a_frag(const _Float16* base, int stride, int lane) {
  const _Float16* rowp = base + (size_t)(lane & 15) * stride;
  const int ks = (lane & 16) ? 16 : 0;
  v16h a;
#pragma unroll
  for (int p = 0; p < 8; ++p) {
    const int kb = ks + 2 * p;
    a[2 * p]     = rowp[kb];
    a[2 * p + 1] = rowp[kb + 1];
  }
  return a;
}

// B fragment: 32x16 f16 from a [n][k]-transposed tile.
__device__ __forceinline__ v16h load_b_frag(const _Float16* baseT, int stride, int lane) {
  const _Float16* colp = baseT + (size_t)(lane & 15) * stride + ((lane & 16) ? 16 : 0);
  v16h b;
#pragma unroll
  for (int e = 0; e < 16; ++e) b[e] = colp[e];
  return b;
}

// ---------------------------------------------------------------------------
// Positional encoding table  pe[t][c]
// ---------------------------------------------------------------------------
__global__ void pe_kernel(float* __restrict__ pe) {
  int i = blockIdx.x * blockDim.x + threadIdx.x;
  if (i >= cT * cH) return;
  int c = i % cH, t = i / cH;
  float c2  = (float)(c & ~1);
  float div = __expf(-9.210340371976184f * c2 / (float)cH);  // ln(10000)
  float ang = (float)t * div;
  pe[i] = (c & 1) ? __cosf(ang) : __sinf(ang);
}

__global__ void cvt_f16_kernel(const float* __restrict__ src, _Float16* __restrict__ dst, int n) {
  int i = blockIdx.x * blockDim.x + threadIdx.x;
  if (i < n) dst[i] = (_Float16)src[i];
}

// ---------------------------------------------------------------------------
// WMMA GEMM: Y[M,N] = epi( A[M,K] @ Wh[K,N] )
// Block tile 128x64, 4 waves, each wave 32x64 (2 A frags x 4 B frags = 8 WMMA
// per K step, B fragments reused across both A fragments, 2-deep B-frag
// pipeline so one LDS round trip stays in flight).
//   A_HALF: A is f16 (else f32, converted during staging)
//   NGUARD: N not a multiple of 64. M % 128 == 0, K % 32 == 0 always.
// ---------------------------------------------------------------------------
template<bool A_HALF, bool NGUARD>
__global__ __launch_bounds__(128) void gemm_wmma_kernel(
    const void* __restrict__ Aptr,
    const _Float16* __restrict__ W, const float* __restrict__ bias,
    const float* __restrict__ Res, const float* __restrict__ PE,
    float* __restrict__ Yf, _Float16* __restrict__ Yh,
    int M, int K, int N, int relu)
{
  const int tid = threadIdx.x, lane = tid & 31, wave = tid >> 5;
  const int bm = blockIdx.y * 128, bn = blockIdx.x * 64;
  const float*    Af = (const float*)Aptr;
  const _Float16* Ah = (const _Float16*)Aptr;
  const v8h vzero = {};

  v8f acc[2][4] = {};

  __shared__ __align__(16) _Float16 sA[2][128][32];  // [buf][row][k]
  __shared__ __align__(16) _Float16 sB[2][64][32];   // [buf][n][k] transposed
  const int nk = K >> 5;

  v4f aF[8];
  v8h aH[4];
  v8h b0, b1;

  auto loadRegs = [&](int kt) {
    const int k0 = kt << 5;
    if constexpr (A_HALF) {
#pragma unroll
      for (int it = 0; it < 4; ++it)
        aH[it] = *(const v8h*)(Ah + (size_t)(bm + (tid >> 2) + 32 * it) * K +
                               k0 + (tid & 3) * 8);
    } else {
#pragma unroll
      for (int it = 0; it < 8; ++it)
        aF[it] = *(const v4f*)(Af + (size_t)(bm + (tid >> 3) + 16 * it) * K +
                               k0 + (tid & 7) * 4);
    }
    if (!NGUARD || bn + (tid & 7) * 8 + 8 <= N) {
      b0 = *(const v8h*)(W + (size_t)(k0 + 2 * (tid >> 3)) * N + bn + (tid & 7) * 8);
      b1 = *(const v8h*)(W + (size_t)(k0 + 2 * (tid >> 3) + 1) * N + bn + (tid & 7) * 8);
    } else {
      b0 = vzero; b1 = vzero;
    }
  };
  auto storeLds = [&](int kt) {
    const int buf = kt & 1;
    if constexpr (A_HALF) {
#pragma unroll
      for (int it = 0; it < 4; ++it)
        *(v8h*)&sA[buf][(tid >> 2) + 32 * it][(tid & 3) * 8] = aH[it];
    } else {
#pragma unroll
      for (int it = 0; it < 8; ++it)
        *(v4h*)&sA[buf][(tid >> 3) + 16 * it][(tid & 7) * 4] =
            __builtin_convertvector(aF[it], v4h);
    }
#pragma unroll
    for (int e = 0; e < 8; ++e) {
      v2h p; p[0] = b0[e]; p[1] = b1[e];
      *(v2h*)&sB[buf][(tid & 7) * 8 + e][2 * (tid >> 3)] = p;
    }
  };

  loadRegs(0);
  for (int kt = 0; kt < nk; ++kt) {
    storeLds(kt);
    __syncthreads();
    if (kt + 1 < nk) loadRegs(kt + 1);
    if (kt + 2 < nk) {
      const int kp2 = (kt + 2) << 5;
      if constexpr (A_HALF)
        __builtin_prefetch(Ah + (size_t)(bm + (tid >> 2)) * K + kp2, 0, 0);
      else
        __builtin_prefetch(Af + (size_t)(bm + (tid >> 3)) * K + kp2, 0, 0);
      __builtin_prefetch(W + (size_t)(kp2 + 2 * (tid >> 3)) * N + bn, 0, 0);
    }
    const int buf = kt & 1;
    v16h af0 = load_a_frag(&sA[buf][32 * wave][0], 32, lane);
    v16h af1 = load_a_frag(&sA[buf][32 * wave + 16][0], 32, lane);
    v16h f0 = load_b_frag(&sB[buf][0][0], 32, lane);
    v16h f1 = load_b_frag(&sB[buf][16][0], 32, lane);
#pragma unroll
    for (int nt = 0; nt < 4; ++nt) {
      v16h cur = f0;
      f0 = f1;
      if (nt + 2 < 4) f1 = load_b_frag(&sB[buf][16 * (nt + 2)][0], 32, lane);
      if (NGUARD && bn + 16 * nt >= N) continue;
      acc[0][nt] = wmma_f16f32(af0, cur, acc[0][nt]);
      acc[1][nt] = wmma_f16f32(af1, cur, acc[1][nt]);
    }
    // no trailing barrier: next store targets the other LDS buffer.
  }

  const int rofs = (lane & 16) ? 8 : 0;
  const int n0 = lane & 15;
#pragma unroll
  for (int half = 0; half < 2; ++half)
#pragma unroll
    for (int nt = 0; nt < 4; ++nt)
#pragma unroll
      for (int r = 0; r < 8; ++r) {
        int gm = bm + 32 * wave + 16 * half + rofs + r;
        int gn = bn + 16 * nt + n0;
        if (NGUARD && gn >= N) continue;
        float v = acc[half][nt][r];
        if (bias) v += bias[gn];
        if (PE)   v += PE[(size_t)(gm & (cT - 1)) * cH + gn];
        if (Res)  v += Res[(size_t)gm * N + gn];
        if (relu) v = fmaxf(v, 0.0f);
        if (Yf) Yf[(size_t)gm * N + gn] = v;
        if (Yh) Yh[(size_t)gm * N + gn] = (_Float16)v;
      }
}

// ---------------------------------------------------------------------------
// Fused Q/K/V GEMM (M=cM, K=N=cH): one staged A tile, three B tiles,
// 12 WMMAs per K step, 2-deep pipelined B-fragment loads.
// Q output pre-scaled by 1/sqrt(dh).
// ---------------------------------------------------------------------------
__global__ __launch_bounds__(128) void qkv_gemm_kernel(
    const float* __restrict__ X,
    const _Float16* __restrict__ Wq, const _Float16* __restrict__ Wk,
    const _Float16* __restrict__ Wv,
    _Float16* __restrict__ Yq, _Float16* __restrict__ Yk,
    _Float16* __restrict__ Yv, float qscale)
{
  constexpr int K = cH, N = cH;
  const int tid = threadIdx.x, lane = tid & 31, wave = tid >> 5;
  const int bm = blockIdx.y * 64, bn = blockIdx.x * 64;

  __shared__ __align__(16) _Float16 sA[2][64][32];
  __shared__ __align__(16) _Float16 sB[2][3][64][32];
  constexpr int nk = K >> 5;

  v4f aF[4];
  v8h bw[3][2];
  v8f acc[3][4] = {};
  const _Float16* Ws[3] = {Wq, Wk, Wv};

  auto loadRegs = [&](int kt) {
    const int k0 = kt << 5;
#pragma unroll
    for (int it = 0; it < 4; ++it)
      aF[it] = *(const v4f*)(X + (size_t)(bm + (tid >> 3) + 16 * it) * K +
                             k0 + (tid & 7) * 4);
#pragma unroll
    for (int w = 0; w < 3; ++w) {
      bw[w][0] = *(const v8h*)(Ws[w] + (size_t)(k0 + 2 * (tid >> 3)) * N + bn + (tid & 7) * 8);
      bw[w][1] = *(const v8h*)(Ws[w] + (size_t)(k0 + 2 * (tid >> 3) + 1) * N + bn + (tid & 7) * 8);
    }
  };
  auto storeLds = [&](int kt) {
    const int buf = kt & 1;
#pragma unroll
    for (int it = 0; it < 4; ++it)
      *(v4h*)&sA[buf][(tid >> 3) + 16 * it][(tid & 7) * 4] =
          __builtin_convertvector(aF[it], v4h);
#pragma unroll
    for (int w = 0; w < 3; ++w)
#pragma unroll
      for (int e = 0; e < 8; ++e) {
        v2h p; p[0] = bw[w][0][e]; p[1] = bw[w][1][e];
        *(v2h*)&sB[buf][w][(tid & 7) * 8 + e][2 * (tid >> 3)] = p;
      }
  };

  loadRegs(0);
  for (int kt = 0; kt < nk; ++kt) {
    storeLds(kt);
    __syncthreads();
    if (kt + 1 < nk) loadRegs(kt + 1);
    if (kt + 2 < nk) {
      const int kp2 = (kt + 2) << 5;
      __builtin_prefetch(X + (size_t)(bm + (tid >> 3)) * K + kp2, 0, 0);
      __builtin_prefetch(Wq + (size_t)(kp2 + 2 * (tid >> 3)) * N + bn, 0, 0);
      __builtin_prefetch(Wk + (size_t)(kp2 + 2 * (tid >> 3)) * N + bn, 0, 0);
      __builtin_prefetch(Wv + (size_t)(kp2 + 2 * (tid >> 3)) * N + bn, 0, 0);
    }
    const int buf = kt & 1;
    v16h af = load_a_frag(&sA[buf][16 * wave][0], 32, lane);
    v16h f0 = load_b_frag(&sB[buf][0][0][0], 32, lane);
    v16h f1 = load_b_frag(&sB[buf][0][16][0], 32, lane);
#pragma unroll
    for (int idx = 0; idx < 12; ++idx) {
      const int w = idx >> 2, nt = idx & 3;
      v16h cur = f0;
      f0 = f1;
      if (idx + 2 < 12) {
        const int w2 = (idx + 2) >> 2, nt2 = (idx + 2) & 3;
        f1 = load_b_frag(&sB[buf][w2][16 * nt2][0], 32, lane);
      }
      acc[w][nt] = wmma_f16f32(af, cur, acc[w][nt]);
    }
  }

  const int rofs = (lane & 16) ? 8 : 0;
  const int n0 = lane & 15;
  _Float16* Ys[3] = {Yq, Yk, Yv};
#pragma unroll
  for (int w = 0; w < 3; ++w) {
    const float sc = (w == 0) ? qscale : 1.0f;
#pragma unroll
    for (int nt = 0; nt < 4; ++nt)
#pragma unroll
      for (int r = 0; r < 8; ++r) {
        int gm = bm + 16 * wave + rofs + r;
        int gn = bn + 16 * nt + n0;
        Ys[w][(size_t)gm * N + gn] = (_Float16)(acc[w][nt][r] * sc);
      }
  }
}

// ---------------------------------------------------------------------------
// accl projection (K=9): plain FMA kernel; W (9x512) stays L2-resident
// ---------------------------------------------------------------------------
__global__ __launch_bounds__(256) void accl_proj_kernel(
    const float* __restrict__ accl, const float* __restrict__ W,
    const float* __restrict__ bias, const float* __restrict__ pe,
    float* __restrict__ Y)
{
  const int m = blockIdx.x;
  const int n = blockIdx.y * 256 + threadIdx.x;
  const int t = m & (cT - 1);
  float a[cA];
#pragma unroll
  for (int j = 0; j < cA; ++j) a[j] = accl[(size_t)m * cA + j];
  float v = bias[n] + pe[(size_t)t * cH + n];
#pragma unroll
  for (int j = 0; j < cA; ++j) v += a[j] * W[j * cH + n];
  Y[(size_t)m * cH + n] = v;
}

// ---------------------------------------------------------------------------
// Depthwise conv (K=3, SAME) on two streams, summed (float4 vectorized)
// ---------------------------------------------------------------------------
__global__ void dwconv_add_kernel(const float* __restrict__ r, const float* __restrict__ a,
                                  const float* __restrict__ kr, const float* __restrict__ ka,
                                  float* __restrict__ x)
{
  size_t i = (size_t)blockIdx.x * blockDim.x + threadIdx.x;   // over cNE/4
  if (i >= cNE / 4) return;
  const int cb = (int)(i % (cH / 4));
  const size_t bt = i / (cH / 4);
  const int t = (int)(bt % cT);
  const size_t rowbase = (bt - (size_t)t) * cH + cb * 4;
  v4f acc = {};
#pragma unroll
  for (int j = 0; j < 3; ++j) {
    int tt = t + j - 1;
    if (tt >= 0 && tt < cT) {
      v4f rv = *(const v4f*)(r + rowbase + (size_t)tt * cH);
      v4f av = *(const v4f*)(a + rowbase + (size_t)tt * cH);
      v4f kv = *(const v4f*)(kr + j * cH + cb * 4);
      v4f k2 = *(const v4f*)(ka + j * cH + cb * 4);
      acc += rv * kv + av * k2;
    }
  }
  *(v4f*)(x + i * 4) = acc;
}

// ---------------------------------------------------------------------------
// Flash attention, one workgroup per (64 q-rows, head, batch); dh=128.
// Q pre-scaled by 1/sqrt(dh).  Q/K/V f16 (B,T,H); O f16 (B,T,H).
// ---------------------------------------------------------------------------
__global__ __launch_bounds__(128) void attn_kernel(const _Float16* __restrict__ Q,
                                                   const _Float16* __restrict__ Kg,
                                                   const _Float16* __restrict__ V,
                                                   _Float16* __restrict__ O)
{
  __shared__ __align__(16) _Float16 sK[32][cDH];      // [key][d]
  __shared__ __align__(16) _Float16 sVt[cDH][32];     // [d][key]
  __shared__ __align__(16) _Float16 sP[4][16][32];    // per-wave P tile
  const int tid = threadIdx.x, lane = tid & 31, wave = tid >> 5;
  const int b = blockIdx.z, h = blockIdx.y;
  const int qbase = blockIdx.x * 64 + wave * 16;
  const _Float16* Qb = Q  + ((size_t)b * cT + qbase) * cH + h * cDH;
  const _Float16* Kb = Kg + (size_t)b * cT * cH + h * cDH;
  const _Float16* Vb = V  + (size_t)b * cT * cH + h * cDH;

  v16h qa[4];
#pragma unroll
  for (int dc = 0; dc < 4; ++dc) qa[dc] = load_a_frag(Qb + dc * 32, cH, lane);

  v8f oacc[8] = {};
  float mrow[8], lrow[8];
#pragma unroll
  for (int r = 0; r < 8; ++r) { mrow[r] = -3.0e38f; lrow[r] = 0.0f; }

  for (int kb = 0; kb < cT; kb += 32) {
#pragma unroll
    for (int i = 0; i < 4; ++i) {               // K tile: b128 in, b128 out
      int idx = tid + 128 * i;
      int row = idx >> 4;
      int col = (idx & 15) << 3;
      *(v8h*)&sK[row][col] = *(const v8h*)(Kb + (size_t)(kb + row) * cH + col);
    }
#pragma unroll
    for (int i = 0; i < 2; ++i) {               // V tile transposed, b32 stores
      int idx = tid + 128 * i;
      int kp  = idx >> 4;
      int col = (idx & 15) << 3;
      v8h v0 = *(const v8h*)(Vb + (size_t)(kb + 2 * kp) * cH + col);
      v8h v1 = *(const v8h*)(Vb + (size_t)(kb + 2 * kp + 1) * cH + col);
#pragma unroll
      for (int e = 0; e < 8; ++e) {
        v2h p; p[0] = v0[e]; p[1] = v1[e];
        *(v2h*)&sVt[col + e][2 * kp] = p;
      }
    }
    __syncthreads();

    // S = Q @ K^T  (two 16-key tiles, chained over d, pipelined B frags)
    v8f s0 = {}, s1 = {};
    {
      v16h bf0 = load_b_frag(&sK[0][0], cDH, lane);
      v16h bf1 = load_b_frag(&sK[16][0], cDH, lane);
#pragma unroll
      for (int dc = 0; dc < 4; ++dc) {
        v16h c0 = bf0, c1 = bf1;
        if (dc < 3) {
          bf0 = load_b_frag(&sK[0][(dc + 1) * 32], cDH, lane);
          bf1 = load_b_frag(&sK[16][(dc + 1) * 32], cDH, lane);
        }
        s0 = wmma_f16f32(qa[dc], c0, s0);
        s1 = wmma_f16f32(qa[dc], c1, s1);
      }
    }

    // online softmax (scale already folded into Q)
    float p0[8], p1[8];
#pragma unroll
    for (int r = 0; r < 8; ++r) {
      float v0 = s0[r], v1 = s1[r];
      float mx = fmaxf(v0, v1);
#pragma unroll
      for (int o2 = 1; o2 < 16; o2 <<= 1) mx = fmaxf(mx, __shfl_xor(mx, o2, 32));
      float mnew = fmaxf(mrow[r], mx);
      float corr = __expf(mrow[r] - mnew);
      float e0 = __expf(v0 - mnew), e1 = __expf(v1 - mnew);
      float rs = e0 + e1;
#pragma unroll
      for (int o2 = 1; o2 < 16; o2 <<= 1) rs += __shfl_xor(rs, o2, 32);
      lrow[r] = lrow[r] * corr + rs;
      mrow[r] = mnew;
#pragma unroll
      for (int nt = 0; nt < 8; ++nt) oacc[nt][r] *= corr;
      p0[r] = e0; p1[r] = e1;
    }

    // restage P (C layout) -> LDS -> A fragment layout
    {
      const int rofs = (lane & 16) ? 8 : 0;
      const int n0 = lane & 15;
#pragma unroll
      for (int r = 0; r < 8; ++r) {
        sP[wave][rofs + r][n0]      = (_Float16)p0[r];
        sP[wave][rofs + r][n0 + 16] = (_Float16)p1[r];
      }
    }
    __syncthreads();

    {
      v16h pa = load_a_frag(&sP[wave][0][0], 32, lane);
      v16h f0 = load_b_frag(&sVt[0][0], 32, lane);
      v16h f1 = load_b_frag(&sVt[16][0], 32, lane);
#pragma unroll
      for (int nt = 0; nt < 8; ++nt) {
        v16h cur = f0;
        f0 = f1;
        if (nt + 2 < 8) f1 = load_b_frag(&sVt[16 * (nt + 2)][0], 32, lane);
        oacc[nt] = wmma_f16f32(pa, cur, oacc[nt]);
      }
    }
    __syncthreads();
  }

  _Float16* Ob = O + ((size_t)b * cT + qbase) * cH + h * cDH;
  const int rofs = (lane & 16) ? 8 : 0;
  const int n0 = lane & 15;
#pragma unroll
  for (int r = 0; r < 8; ++r) {
    float inv = 1.0f / lrow[r];
#pragma unroll
    for (int nt = 0; nt < 8; ++nt)
      Ob[(size_t)(rofs + r) * cH + nt * 16 + n0] = (_Float16)(oacc[nt][r] * inv);
  }
}

// ---------------------------------------------------------------------------
// LayerNorm: one wave32 per 512-element row, v4f vectorized, row kept in regs
// ---------------------------------------------------------------------------
__global__ __launch_bounds__(256) void ln_kernel(const float* __restrict__ X,
                                                 const float* __restrict__ g,
                                                 const float* __restrict__ b,
                                                 float* __restrict__ Y)
{
  const int lane = threadIdx.x & 31, wave = threadIdx.x >> 5;
  const size_t row = (size_t)blockIdx.x * 8 + wave;
  const v4f* xr4 = (const v4f*)(X + row * cH);
  v4f vals[4];
  float s = 0.0f, s2 = 0.0f;
#pragma unroll
  for (int it = 0; it < 4; ++it) {
    v4f v = xr4[lane + 32 * it];
    vals[it] = v;
#pragma unroll
    for (int e = 0; e < 4; ++e) { s += v[e]; s2 += v[e] * v[e]; }
  }
#pragma unroll
  for (int o2 = 1; o2 < 32; o2 <<= 1) { s += __shfl_xor(s, o2, 32); s2 += __shfl_xor(s2, o2, 32); }
  const float mean = s * (1.0f / cH);
  const float var  = s2 * (1.0f / cH) - mean * mean;
  const float inv  = rsqrtf(var + 1e-5f);
  v4f* yr4 = (v4f*)(Y + row * cH);
#pragma unroll
  for (int it = 0; it < 4; ++it) {
    const int c4 = lane + 32 * it;
    v4f gg = *(const v4f*)(g + c4 * 4);
    v4f bb = *(const v4f*)(b + c4 * 4);
    v4f v = vals[it];
#pragma unroll
    for (int e = 0; e < 4; ++e) v[e] = (v[e] - mean) * inv * gg[e] + bb[e];
    yr4[c4] = v;
  }
}

// ---------------------------------------------------------------------------
// Viterbi: one wave32 per batch, state-per-lane (L=16), byte backpointers
// ---------------------------------------------------------------------------
__global__ __launch_bounds__(32) void viterbi_kernel(const float* __restrict__ em,
                                                     const float* __restrict__ st,
                                                     const float* __restrict__ et,
                                                     const float* __restrict__ trans,
                                                     unsigned char* __restrict__ hist,
                                                     long long* __restrict__ out)
{
  const int b = blockIdx.x;
  const int lane = threadIdx.x;
  const int j = lane & 15;
  float tr[cL];
#pragma unroll
  for (int i = 0; i < cL; ++i) tr[i] = trans[i * cL + j];
  const float* emb = em + (size_t)b * cT * cL;
  float score = st[j] + emb[j];
  for (int t = 1; t < cT; ++t) {
    float best = -3.0e38f;
    int arg = 0;
#pragma unroll
    for (int i = 0; i < cL; ++i) {
      float si = __shfl(score, i, 32);
      float tot = si + tr[i];
      if (tot > best) { best = tot; arg = i; }
    }
    score = best + emb[t * cL + j];
    if (lane < cL) hist[((size_t)b * cT + t) * cL + j] = (unsigned char)arg;
  }
  float fin = score + et[j];
  int bj = j;
#pragma unroll
  for (int o2 = 1; o2 < 16; o2 <<= 1) {
    float of = __shfl_xor(fin, o2, 32);
    int   oj = __shfl_xor(bj, o2, 32);
    if (of > fin || (of == fin && oj < bj)) { fin = of; bj = oj; }
  }
  if (lane == 0) {
    int tag = bj;
    out[(size_t)b * cT + (cT - 1)] = (long long)tag;
    for (int t = cT - 1; t >= 1; --t) {
      tag = hist[((size_t)b * cT + t) * cL + tag];
      out[(size_t)b * cT + (t - 1)] = (long long)tag;
    }
  }
}

// ---------------------------------------------------------------------------
// Host-side orchestration
// ---------------------------------------------------------------------------
extern "C" void kernel_launch(void* const* d_in, const int* in_sizes, int n_in,
                              void* d_out, int out_size, void* d_ws, size_t ws_size,
                              hipStream_t stream)
{
  (void)in_sizes; (void)n_in; (void)out_size; (void)ws_size;

  const float* rssi   = (const float*)d_in[0];
  const float* accl   = (const float*)d_in[1];
  const float* rssi_W = (const float*)d_in[2];
  const float* rssi_b = (const float*)d_in[3];
  const float* accl_W = (const float*)d_in[4];
  const float* accl_b = (const float*)d_in[5];
  const float* conv_r = (const float*)d_in[6];
  const float* conv_a = (const float*)d_in[7];
  const float* Wq = (const float*)d_in[8];
  const float* Wk = (const float*)d_in[9];
  const float* Wv = (const float*)d_in[10];
  const float* Wo = (const float*)d_in[11];
  const float* ln1_g = (const float*)d_in[12];
  const float* ln1_b = (const float*)d_in[13];
  const float* W1 = (const float*)d_in[14];
  const float* b1 = (const float*)d_in[15];
  const float* W2 = (const float*)d_in[16];
  const float* b2 = (const float*)d_in[17];
  const float* ln2_g = (const float*)d_in[18];
  const float* ln2_b = (const float*)d_in[19];
  const float* loc_W = (const float*)d_in[20];
  const float* loc_b = (const float*)d_in[21];
  const float* start_trans = (const float*)d_in[22];
  const float* end_trans   = (const float*)d_in[23];
  const float* trans       = (const float*)d_in[24];

  char* ws = (char*)d_ws;
  size_t off = 0;
  auto alloc = [&](size_t bytes) -> void* {
    off = (off + 255) & ~(size_t)255;
    void* p = (void*)(ws + off);
    off += bytes;
    return p;
  };

  float*    pe      = (float*)   alloc((size_t)cT * cH * 4);
  _Float16* wh_rssi = (_Float16*)alloc((size_t)cR * cH * 2);
  _Float16* wh_q    = (_Float16*)alloc((size_t)cH * cH * 2);
  _Float16* wh_k    = (_Float16*)alloc((size_t)cH * cH * 2);
  _Float16* wh_v    = (_Float16*)alloc((size_t)cH * cH * 2);
  _Float16* wh_o    = (_Float16*)alloc((size_t)cH * cH * 2);
  _Float16* wh_1    = (_Float16*)alloc((size_t)cH * cF * 2);
  _Float16* wh_2    = (_Float16*)alloc((size_t)cF * cH * 2);
  _Float16* wh_l    = (_Float16*)alloc((size_t)cH * cL * 2);
  float*    bufR    = (float*)   alloc(cNE * 4);
  float*    bufA    = (float*)   alloc(cNE * 4);
  float*    bufX    = (float*)   alloc(cNE * 4);
  float*    bufT    = (float*)   alloc(cNE * 4);
  float*    bufX1   = (float*)   alloc(cNE * 4);
  _Float16* qh      = (_Float16*)alloc(cNE * 2);
  _Float16* kh      = (_Float16*)alloc(cNE * 2);
  _Float16* vh      = (_Float16*)alloc(cNE * 2);
  _Float16* oh      = (_Float16*)alloc(cNE * 2);
  _Float16* hh      = (_Float16*)alloc((size_t)cM * cF * 2);
  float*    em      = (float*)   alloc((size_t)cM * cL * 4);
  unsigned char* hist = (unsigned char*)alloc((size_t)cM * cL);
  float* bufX2 = bufR;   // bufR is dead after the conv; reuse for x2

  pe_kernel<<<(cT * cH + 255) / 256, 256, 0, stream>>>(pe);
  auto cvt = [&](const float* s, _Float16* d, int n) {
    cvt_f16_kernel<<<(n + 255) / 256, 256, 0, stream>>>(s, d, n);
  };
  cvt(rssi_W, wh_rssi, cR * cH);
  cvt(Wq, wh_q, cH * cH);
  cvt(Wk, wh_k, cH * cH);
  cvt(Wv, wh_v, cH * cH);
  cvt(Wo, wh_o, cH * cH);
  cvt(W1, wh_1, cH * cF);
  cvt(W2, wh_2, cF * cH);
  cvt(loc_W, wh_l, cH * cL);

  dim3 gH(cH / 64, cM / 128);     // N=512 GEMMs (128-row tiles)
  dim3 gF(cF / 64, cM / 128);     // N=2048 GEMM
  dim3 gL(1, cM / 128);           // N=16 GEMM (n-guarded)
  dim3 gQ(cH / 64, cM / 64);      // fused QKV (64-row tiles)

  // r = rssi@W + b + pe  (K=256) ; a = accl@W + b + pe (K=9, FMA kernel)
  gemm_wmma_kernel<false, false><<<gH, 128, 0, stream>>>(rssi, wh_rssi, rssi_b,
      nullptr, pe, bufR, nullptr, cM, cR, cH, 0);
  accl_proj_kernel<<<dim3(cM, cH / 256), 256, 0, stream>>>(accl, accl_W, accl_b,
      pe, bufA);
  // x = dwconv(r) + dwconv(a)
  dwconv_add_kernel<<<(unsigned)((cNE / 4 + 255) / 256), 256, 0, stream>>>(bufR, bufA,
      conv_r, conv_a, bufX);
  // fused q,k,v (f16 outputs, q pre-scaled by 1/sqrt(dh))
  qkv_gemm_kernel<<<gQ, 128, 0, stream>>>(bufX, wh_q, wh_k, wh_v, qh, kh, vh,
      0.088388347648318447f);
  // attention (f16 output)
  dim3 ga(cT / 64, cHEADS, cB);
  attn_kernel<<<ga, 128, 0, stream>>>(qh, kh, vh, oh);
  // x + o@Wo (f16 A operand), then LN1
  gemm_wmma_kernel<true, false><<<gH, 128, 0, stream>>>(oh, wh_o, nullptr,
      bufX, nullptr, bufT, nullptr, cM, cH, cH, 0);
  ln_kernel<<<cM / 8, 256, 0, stream>>>(bufT, ln1_g, ln1_b, bufX1);
  // FFN (hidden in f16), residual, LN2
  gemm_wmma_kernel<false, false><<<gF, 128, 0, stream>>>(bufX1, wh_1, b1,
      nullptr, nullptr, nullptr, hh, cM, cH, cF, 1);
  gemm_wmma_kernel<true, false><<<gH, 128, 0, stream>>>(hh, wh_2, b2,
      bufX1, nullptr, bufT, nullptr, cM, cF, cH, 0);
  ln_kernel<<<cM / 8, 256, 0, stream>>>(bufT, ln2_g, ln2_b, bufX2);
  // emissions (N=16, n-guarded fast path)
  gemm_wmma_kernel<false, true><<<gL, 128, 0, stream>>>(bufX2, wh_l, loc_b,
      nullptr, nullptr, em, nullptr, cM, cH, cL, 0);
  // Viterbi decode -> int64 tags
  viterbi_kernel<<<cB, 32, 0, stream>>>(em, start_trans, end_trans, trans, hist,
      (long long*)d_out);
}
